// MultiHeadAttentionBlock_58471684767951
// MI455X (gfx1250) — matrix operand order repaired and checked by reading, not compiled, not run on hardware
//
#include <hip/hip_runtime.h>

typedef __attribute__((ext_vector_type(16))) _Float16     v16h;
typedef __attribute__((ext_vector_type(8)))  float        v8f;
typedef __attribute__((ext_vector_type(4)))  unsigned int u32x4;
typedef __attribute__((ext_vector_type(8)))  int          i32x8;
typedef __attribute__((ext_vector_type(4)))  int          i32x4;

constexpr int Bc = 4, Lc = 2048, Ec = 1024, Hc = 16, HDc = 64;
constexpr int Mc = Bc * Lc;   // 8192 rows of x / attention output

union FragH { v16h h; unsigned int u[8]; };

// low 32 bits of a generic pointer to LDS == wave-relative LDS byte address
__device__ __forceinline__ unsigned lds_u32(const void* p) {
    return (unsigned)(unsigned long long)p;
}

// ---------------------------------------------------------------- TDM helper
// 2D tensor tile load: tile_dim0=32 halfs (x), tile_dim1=128 rows, data_size=2B,
// LDS padding: 4 dwords (8 halfs) after every 16 dwords (32 halfs) -> 40-half rows.
// D# packing per CDNA5 ISA (groups 2/3 zero => 2D).
// This toolchain exposes the 6-arg builtin:
//   (u32x4 g0, i32x8 g1, i32x4 g2, i32x4 g3, i32x8 extra, i32 cpol)
__device__ __forceinline__ void tdm_issue_tile32x128(const _Float16* g, size_t elem_off,
                                                     unsigned lds_byte,
                                                     unsigned tensor_d0, unsigned tensor_d1)
{
    unsigned long long ga = (unsigned long long)g + elem_off * 2ull;
    u32x4 g0;
    g0[0] = 1u;                                   // count=1, user descriptor
    g0[1] = lds_byte;                             // lds_addr
    g0[2] = (unsigned)(ga & 0xffffffffull);       // global_addr[31:0]
    g0[3] = (unsigned)(ga >> 32) | 0x80000000u;   // global_addr[56:32] | type=2
    i32x8 g1;
    g1[0] = (int)((1u << 16)                      // data_size = 2 bytes
                | (1u << 20)                      // pad_enable
                | (3u << 22)                      // pad_interval: 16 dwords
                | (3u << 25));                    // pad_amount:   4 dwords
    g1[1] = (int)(tensor_d0 << 16);               // tensor_dim0[15:0] @ bits[63:48]
    g1[2] = (int)((tensor_d0 >> 16) | (tensor_d1 << 16));
    g1[3] = (int)((tensor_d1 >> 16) | (32u << 16));   // tile_dim0 = 32
    g1[4] = (int)128;                             // tile_dim1 = 128, tile_dim2 = 0
    g1[5] = (int)tensor_d0;                       // tensor_dim0_stride[31:0]
    g1[6] = 0;
    g1[7] = 0;
    i32x4 z4 = {0, 0, 0, 0};
    i32x8 z8 = {0, 0, 0, 0, 0, 0, 0, 0};
    __builtin_amdgcn_tensor_load_to_lds(g0, g1, z4, z4, z8, 0);
}

// ---------------------------------------------------------------- convert
__global__ void cvt_f32_f16(const float* __restrict__ in, _Float16* __restrict__ out, int n) {
    int i = blockIdx.x * blockDim.x + threadIdx.x;
    int stride = gridDim.x * blockDim.x;
    for (; i < n; i += stride) out[i] = (_Float16)in[i];
}

// ---------------------------------------------------------------- GEMM
// C[M,N] = A[M,K](f16) * W[N,K](f16)^T + bias, 128x128 tile per block,
// 8 wave32s, each wave a 32x64 subtile = 2x4 wmma accumulators.
// Tiles staged by the Tensor Data Mover, double-buffered in LDS.
// outh != null -> f16 head-scattered [B][H][L][HD] (QKV path, with scale)
// outf != null -> f32 [M][N]                       (output projection)
__global__ __launch_bounds__(256)
void gemm_wmma_f16(const _Float16* __restrict__ A,
                   const _Float16* __restrict__ W,
                   const float* __restrict__ bias,
                   _Float16* __restrict__ outh,
                   float* __restrict__ outf,
                   float scale)
{
    constexpr int K = Ec, N = Ec;
    constexpr int NK = K / 32;
    __shared__ _Float16 sA[2][128][40];   // [buf][row][k], 80 B rows (TDM-padded)
    __shared__ _Float16 sB[2][128][40];   // [buf][n][k] == W rows (B^T layout)

    const int t    = threadIdx.x;
    const int wave = t >> 5, lane = t & 31;
    const int lrow = lane & 15, half = lane >> 4;
    const int wm   = wave & 3, wn = wave >> 2;
    const int mbase = blockIdx.y * 128;
    const int nbase = blockIdx.x * 128;

    const unsigned aoff[2] = { lds_u32(&sA[0][0][0]), lds_u32(&sA[1][0][0]) };
    const unsigned boff[2] = { lds_u32(&sB[0][0][0]), lds_u32(&sB[1][0][0]) };

    v8f acc[2][4] = {};

    if (wave == 0) {      // prime the pipeline: tile 0 -> buffer 0 (2 TDM ops)
        tdm_issue_tile32x128(A, (size_t)mbase * K, aoff[0], K, (unsigned)Mc);
        tdm_issue_tile32x128(W, (size_t)nbase * K, boff[0], K, (unsigned)N);
    }

    for (int it = 0; it < NK; ++it) {
        const int cur = it & 1;
        if (wave == 0) {
            if (it + 1 < NK) {   // issue next tile, then retire current (in-order)
                int kb = (it + 1) * 32;
                tdm_issue_tile32x128(A, (size_t)mbase * K + kb, aoff[cur ^ 1], K, (unsigned)Mc);
                tdm_issue_tile32x128(W, (size_t)nbase * K + kb, boff[cur ^ 1], K, (unsigned)N);
                __builtin_amdgcn_s_wait_tensorcnt(2);
            } else {
                __builtin_amdgcn_s_wait_tensorcnt(0);
            }
        }
        __syncthreads();                       // current buffer visible to all waves

        FragH af[2], bf[4];
        #pragma unroll
        for (int f = 0; f < 2; ++f) {
            int row = wm * 32 + f * 16 + lrow;
            #pragma unroll
            for (int r = 0; r < 8; ++r) {      // A layout: VGPR r -> K pair
                int k = ((r & 3) << 1) + (half << 3) + ((r >> 2) << 4);
                af[f].u[r] = *(const unsigned int*)&sA[cur][row][k];
            }
        }
        #pragma unroll
        for (int g = 0; g < 4; ++g) {
            int col = wn * 64 + g * 16 + lrow;
            #pragma unroll
            for (int r = 0; r < 8; ++r) {      // B layout: lane=N col, VGPR r -> K pair
                int k = ((r & 3) << 1) + (half << 3) + ((r >> 2) << 4);
                bf[g].u[r] = *(const unsigned int*)&sB[cur][col][k];
            }
        }
        #pragma unroll
        for (int f = 0; f < 2; ++f)
            #pragma unroll
            for (int g = 0; g < 4; ++g)
                acc[f][g] = __builtin_amdgcn_wmma_f32_16x16x32_f16(
                    false, af[f].h, false, bf[g].h, (short)0, acc[f][g], false, false);

        __syncthreads();                       // all reads of current buffer done
    }

    #pragma unroll
    for (int f = 0; f < 2; ++f) {
        #pragma unroll
        for (int g = 0; g < 4; ++g) {
            unsigned col = nbase + wn * 64 + g * 16 + lrow;
            float bv = bias[col];
            #pragma unroll
            for (int r = 0; r < 8; ++r) {
                unsigned row = mbase + wm * 32 + f * 16 + r + (half << 3);
                float val = (acc[f][g][r] + bv) * scale;
                if (outh) {
                    unsigned b = row >> 11, lr = row & (Lc - 1);    // Lc = 2048
                    unsigned h = col >> 6,  d  = col & (HDc - 1);   // HDc = 64
                    outh[((size_t)(b * Hc + h) * Lc + lr) * HDc + d] = (_Float16)val;
                } else {
                    outf[(size_t)row * N + col] = val;
                }
            }
        }
    }
}

// ---------------------------------------------------------------- attention
// Flash attention; block = 8 waves x 16 query rows = 128 rows per (b,h).
__global__ __launch_bounds__(256)
void attn_wmma(const _Float16* __restrict__ q, const _Float16* __restrict__ k,
               const _Float16* __restrict__ v, const int* __restrict__ mask,
               _Float16* __restrict__ out)
{
    __shared__ _Float16 sK[64][72];        // [key][d]  row-major
    __shared__ _Float16 sVt[64][72];       // [d][key]  transposed for P*V B frags
    __shared__ int      sMask[64];
    __shared__ _Float16 sP[8][16][72];     // per-wave P staging (C->A relayout)

    const int t    = threadIdx.x;
    const int wave = t >> 5, lane = t & 31;
    const int lrow = lane & 15, half = lane >> 4;
    const int hb = blockIdx.y, bb = blockIdx.z;
    const int qbase = blockIdx.x * 128;

    const size_t headoff = (size_t)(bb * Hc + hb) * Lc * HDc;
    const _Float16* qh = q + headoff;
    const _Float16* kh = k + headoff;
    const _Float16* vh = v + headoff;
    const int* mb = mask + bb * Lc;

    // Q strip (16 rows x 64 d) resident as two A fragments
    FragH qa[2];
    {
        int row = qbase + wave * 16 + lrow;
        #pragma unroll
        for (int f = 0; f < 2; ++f)
            #pragma unroll
            for (int r = 0; r < 8; ++r) {
                int d = f * 32 + ((r & 3) << 1) + (half << 3) + ((r >> 2) << 4);
                qa[f].u[r] = *(const unsigned int*)&qh[(size_t)row * HDc + d];
            }
    }

    v8f o[4] = {};
    float mrow[8], lsum[8];
    #pragma unroll
    for (int r = 0; r < 8; ++r) { mrow[r] = -3.0e38f; lsum[r] = 0.f; }

    for (int kt = 0; kt < Lc; kt += 64) {
        __syncthreads();
        #pragma unroll
        for (int i = 0; i < 8; ++i) {
            int j   = t + i * 256;
            int key = j >> 5;
            int c2  = (j & 31) << 1;
            unsigned int kv = *(const unsigned int*)&kh[(size_t)(kt + key) * HDc + c2];
            *(unsigned int*)&sK[key][c2] = kv;
            unsigned int vv = *(const unsigned int*)&vh[(size_t)(kt + key) * HDc + c2];
            sVt[c2][key]     = ((const _Float16*)&vv)[0];
            sVt[c2 + 1][key] = ((const _Float16*)&vv)[1];
        }
        if (t < 64) sMask[t] = mb[kt + t];
        __syncthreads();

        // S = Q * K^T : 16 rows x 64 keys per wave (4 column frags, 2 K-steps)
        v8f s[4] = {};
        #pragma unroll
        for (int g = 0; g < 4; ++g) {
            FragH b0, b1;
            int key = g * 16 + lrow;
            #pragma unroll
            for (int r = 0; r < 8; ++r) {
                int d = ((r & 3) << 1) + (half << 3) + ((r >> 2) << 4);
                b0.u[r] = *(const unsigned int*)&sK[key][d];
                b1.u[r] = *(const unsigned int*)&sK[key][d + 32];
            }
            s[g] = __builtin_amdgcn_wmma_f32_16x16x32_f16(
                false, qa[0].h, false, b0.h, (short)0, s[g], false, false);
            s[g] = __builtin_amdgcn_wmma_f32_16x16x32_f16(
                false, qa[1].h, false, b1.h, (short)0, s[g], false, false);
        }

        // mask: each lane owns one key column -> uniform per lane
        #pragma unroll
        for (int g = 0; g < 4; ++g)
            if (sMask[g * 16 + lrow] == 0) {
                #pragma unroll
                for (int r = 0; r < 8; ++r) s[g][r] = -1.0e10f;
            }

        // online softmax over rows (row = r + 8*half, spread over 16 lanes)
        #pragma unroll
        for (int r = 0; r < 8; ++r) {
            float vmax = fmaxf(fmaxf(s[0][r], s[1][r]), fmaxf(s[2][r], s[3][r]));
            #pragma unroll
            for (int off = 1; off < 16; off <<= 1)
                vmax = fmaxf(vmax, __shfl_xor(vmax, off, 32));
            float mn = fmaxf(mrow[r], vmax);
            float sc = expf(mrow[r] - mn);      // 0 on first tile
            float psum = 0.f;
            #pragma unroll
            for (int g = 0; g < 4; ++g) {
                float p = expf(s[g][r] - mn);
                s[g][r] = p;
                psum += p;
            }
            #pragma unroll
            for (int off = 1; off < 16; off <<= 1)
                psum += __shfl_xor(psum, off, 32);
            lsum[r] = lsum[r] * sc + psum;
            mrow[r] = mn;
            #pragma unroll
            for (int g = 0; g < 4; ++g) o[g][r] *= sc;
        }

        // relayout P: f32 C-frag -> f16 A-frag via per-wave LDS staging
        #pragma unroll
        for (int g = 0; g < 4; ++g)
            #pragma unroll
            for (int r = 0; r < 8; ++r)
                sP[wave][r + (half << 3)][g * 16 + lrow] = (_Float16)s[g][r];

        asm volatile("s_wait_dscnt 0" ::: "memory");  // intra-wave cross-lane LDS RAW

        FragH pa[2];
        #pragma unroll
        for (int f = 0; f < 2; ++f)
            #pragma unroll
            for (int r = 0; r < 8; ++r) {
                int kk = f * 32 + ((r & 3) << 1) + (half << 3) + ((r >> 2) << 4);
                pa[f].u[r] = *(const unsigned int*)&sP[wave][lrow][kk];
            }

        // O += P * V  (V staged transposed: contiguous key pairs per d column)
        #pragma unroll
        for (int g = 0; g < 4; ++g) {
            FragH b0, b1;
            int d = g * 16 + lrow;
            #pragma unroll
            for (int r = 0; r < 8; ++r) {
                int kk = ((r & 3) << 1) + (half << 3) + ((r >> 2) << 4);
                b0.u[r] = *(const unsigned int*)&sVt[d][kk];
                b1.u[r] = *(const unsigned int*)&sVt[d][kk + 32];
            }
            o[g] = __builtin_amdgcn_wmma_f32_16x16x32_f16(
                false, pa[0].h, false, b0.h, (short)0, o[g], false, false);
            o[g] = __builtin_amdgcn_wmma_f32_16x16x32_f16(
                false, pa[1].h, false, b1.h, (short)0, o[g], false, false);
        }
    }

    // normalize and write [B][L][E] with heads merged (E index = h*HD + d)
    #pragma unroll
    for (int g = 0; g < 4; ++g) {
        int d = g * 16 + lrow;
        #pragma unroll
        for (int r = 0; r < 8; ++r) {
            int row = qbase + wave * 16 + r + (half << 3);
            float val = o[g][r] / lsum[r];
            out[(size_t)(bb * Lc + row) * Ec + hb * HDc + d] = (_Float16)val;
        }
    }
}

// ---------------------------------------------------------------- launch
extern "C" void kernel_launch(void* const* d_in, const int* in_sizes, int n_in,
                              void* d_out, int out_size, void* d_ws, size_t ws_size,
                              hipStream_t stream) {
    (void)in_sizes; (void)n_in; (void)out_size; (void)ws_size;
    const float* x  = (const float*)d_in[0];
    const int*   mk = (const int*)d_in[1];
    const float* Wq = (const float*)d_in[2];
    const float* bq = (const float*)d_in[3];
    const float* Wk = (const float*)d_in[4];
    const float* bk = (const float*)d_in[5];
    const float* Wv = (const float*)d_in[6];
    const float* bv = (const float*)d_in[7];
    const float* Wo = (const float*)d_in[8];
    const float* bo = (const float*)d_in[9];
    float* out = (float*)d_out;

    char* ws = (char*)d_ws;
    const size_t MB = 1ull << 20;
    _Float16* x16  = (_Float16*)(ws + 0);         // 16 MB, reused for attention out
    _Float16* wq16 = (_Float16*)(ws + 16 * MB);   // 2 MB each
    _Float16* wk16 = (_Float16*)(ws + 18 * MB);
    _Float16* wv16 = (_Float16*)(ws + 20 * MB);
    _Float16* wo16 = (_Float16*)(ws + 22 * MB);
    _Float16* q16  = (_Float16*)(ws + 24 * MB);   // 16 MB each
    _Float16* k16  = (_Float16*)(ws + 40 * MB);
    _Float16* v16  = (_Float16*)(ws + 56 * MB);
    _Float16* a16  = x16;                         // x dead after QKV projections

    const int nx = Mc * Ec;
    const int nw = Ec * Ec;
    cvt_f32_f16<<<2048, 256, 0, stream>>>(x,  x16,  nx);
    cvt_f32_f16<<<1024, 256, 0, stream>>>(Wq, wq16, nw);
    cvt_f32_f16<<<1024, 256, 0, stream>>>(Wk, wk16, nw);
    cvt_f32_f16<<<1024, 256, 0, stream>>>(Wv, wv16, nw);
    cvt_f32_f16<<<1024, 256, 0, stream>>>(Wo, wo16, nw);

    dim3 gg(Ec / 128, Mc / 128);   // (8, 64)
    gemm_wmma_f16<<<gg, 256, 0, stream>>>(x16, wq16, bq, q16, nullptr, 0.125f); // fold 1/sqrt(HD)
    gemm_wmma_f16<<<gg, 256, 0, stream>>>(x16, wk16, bk, k16, nullptr, 1.0f);
    gemm_wmma_f16<<<gg, 256, 0, stream>>>(x16, wv16, bv, v16, nullptr, 1.0f);

    attn_wmma<<<dim3(Lc / 128, Hc, Bc), 256, 0, stream>>>(q16, k16, v16, mk, a16);

    gemm_wmma_f16<<<gg, 256, 0, stream>>>(a16, wo16, bo, nullptr, out, 1.0f);
}
